// MultiHeadAttention_84567906058706
// MI455X (gfx1250) — matrix operand order repaired and checked by reading, not compiled, not run on hardware
//
#include <hip/hip_runtime.h>
#include <hip/hip_bf16.h>

// ---------------------------------------------------------------------------
// MHA forward for MI455X (gfx1250, wave32, WMMA 16x16x32 bf16 -> f32 acc)
// B=4, S=2048, D=1024, H=16, hd=64
// ---------------------------------------------------------------------------

typedef __bf16 bf16_t;
typedef __attribute__((ext_vector_type(16))) __bf16 v16bf;
typedef __attribute__((ext_vector_type(8)))  __bf16 v8bf;
typedef __attribute__((ext_vector_type(4)))  __bf16 v4bf;
typedef __attribute__((ext_vector_type(8)))  float  v8f;
typedef __attribute__((ext_vector_type(4)))  float  v4f;
typedef __attribute__((ext_vector_type(4)))  unsigned int v4u;
typedef __attribute__((ext_vector_type(8)))  int    v8i;
typedef __attribute__((ext_vector_type(4)))  int    v4i;

#define MODEL_D 1024
#define SEQ     2048
#define BATCH   4
#define HEADS   16
#define HDIM    64
#define MROWS   (BATCH * SEQ)   // 8192

// 1/sqrt(hd) * log2(e): folded into the q projection so softmax runs in
// base-2 domain with no extra multiplies.
#define QSCALE  (0.125f * 1.44269504088896f)

static __device__ __forceinline__ v16bf cat8(v8bf lo, v8bf hi) {
  v16bf r;
#pragma unroll
  for (int i = 0; i < 8; ++i) { r[i] = lo[i]; r[i + 8] = hi[i]; }
  return r;
}

static __device__ __forceinline__ v8f zero8() {
  v8f z;
#pragma unroll
  for (int i = 0; i < 8; ++i) z[i] = 0.0f;
  return z;
}

static __device__ __forceinline__ v8f wmma_bf16(v16bf a, v16bf b, v8f c) {
  // (neg_a, A, neg_b, B, c_mod, C, reuse_a, reuse_b)
  return __builtin_amdgcn_wmma_f32_16x16x32_bf16(false, a, false, b,
                                                 (short)0, c, false, false);
}

// Raw v_max_num_f32 (no canonicalize insertion).
static __device__ __forceinline__ float fmax_raw(float a, float b) {
  float d;
  asm("v_max_num_f32_e32 %0, %1, %2" : "=v"(d) : "v"(a), "v"(b));
  return d;
}

// Butterfly max across each 16-lane row: single fused v_max_num_f32_dpp
// per step (DPP on src0), no mov/canonicalize overhead.
static __device__ __forceinline__ float rowmax16(float x) {
  float t;
  asm("v_max_num_f32_dpp %0, %1, %1 row_xmask:1 row_mask:0xf bank_mask:0xf bound_ctrl:1"
      : "=&v"(t) : "v"(x));
  x = t;
  asm("v_max_num_f32_dpp %0, %1, %1 row_xmask:2 row_mask:0xf bank_mask:0xf bound_ctrl:1"
      : "=&v"(t) : "v"(x));
  x = t;
  asm("v_max_num_f32_dpp %0, %1, %1 row_xmask:4 row_mask:0xf bank_mask:0xf bound_ctrl:1"
      : "=&v"(t) : "v"(x));
  x = t;
  asm("v_max_num_f32_dpp %0, %1, %1 row_xmask:8 row_mask:0xf bank_mask:0xf bound_ctrl:1"
      : "=&v"(t) : "v"(x));
  return t;
}

#if __has_builtin(__builtin_amdgcn_tensor_load_to_lds)
#define USE_TDM 1
// Build a 2D-tile Tensor DMA descriptor (D#) per CDNA5 ISA 8.3/8.4 and issue
// TENSOR_LOAD_TO_LDS (6-arg clang-23 form: g0, g1, g2, g3, extra, cpol).
// dims/strides in elements (data_size = 2 bytes).
// padInterval: pad after 2^(pi+1) dwords; padAmount: (pa+1) dwords.
static __device__ __forceinline__ void tdm_load_2d(
    unsigned lds_off, const void* gptr, unsigned tileW, unsigned tileH,
    unsigned rowStride, unsigned padInterval, unsigned padAmount) {
  const unsigned long long ga = (unsigned long long)gptr;
  v4u g0;
  g0[0] = 1u;                                   // count=1, user descriptor
  g0[1] = lds_off;                              // lds_addr (bytes)
  g0[2] = (unsigned)(ga & 0xFFFFFFFFu);         // global_addr[31:0]
  g0[3] = (unsigned)((ga >> 32) & 0x01FFFFFFu)  // global_addr[56:32]
          | (2u << 30);                         // type = 2 ("image")
  v8i g1;
  g1[0] = (int)((1u << 16)                      // data_size = 1 (2 bytes)
                | (1u << 20)                    // pad_enable
                | (padInterval << 22)           // pad_interval
                | (padAmount << 25));           // pad_amount
  g1[1] = (int)(tileW << 16);                   // tensor_dim0[15:0]
  g1[2] = (int)(tileH << 16);                   // tensor_dim1[15:0]
  g1[3] = (int)(tileW << 16);                   // tile_dim0
  g1[4] = (int)tileH;                           // tile_dim1 (tile_dim2 = 0)
  g1[5] = (int)rowStride;                       // tensor_dim0_stride[31:0]
  g1[6] = 0;
  g1[7] = 0;
  v4i z4 = {0, 0, 0, 0};                        // groups 2/3 unused (2D)
  v8i z8 = {0, 0, 0, 0, 0, 0, 0, 0};            // extra descriptor words
  __builtin_amdgcn_tensor_load_to_lds(g0, g1, z4, z4, z8, 0);
}
#else
#define USE_TDM 0
#endif

// ---------------------------------------------------------------------------
// fp32 -> bf16 weight conversion (vectorized x4)
// ---------------------------------------------------------------------------
__global__ __launch_bounds__(256) void cvt_f32_bf16_kernel(
    const float* __restrict__ src, bf16_t* __restrict__ dst, int n4) {
  int i = blockIdx.x * 256 + threadIdx.x;
  if (i < n4) {
    v4f f = ((const v4f*)src)[i];
    v4bf b;
#pragma unroll
    for (int j = 0; j < 4; ++j) b[j] = (bf16_t)f[j];
    ((v4bf*)dst)[i] = b;
  }
}

// ---------------------------------------------------------------------------
// Tiled GEMM:  Out = (A[M,K] @ W[N,K]^T + bias[N]) * outScale
//   block tile 128x128, 256 threads = 8 waves, wave tile 64(M) x 32(N)
//   OUT_MODE: 0 = bf16 row-major [M,N]; 1 = fp32 row-major [M,N];
//             2 = bf16 column-major [N,M] (vectorized stores; feeds flash V^T)
// ---------------------------------------------------------------------------
template <bool A_F32, int OUT_MODE>
__global__ __launch_bounds__(256) void gemm_wmma_kernel(
    const void* __restrict__ Ap, const bf16_t* __restrict__ W,
    const float* __restrict__ bias, void* __restrict__ Outp,
    int M, int N, int K, float outScale) {
  __shared__ bf16_t As[128][40];   // +8 pad, rows 16B aligned
  __shared__ bf16_t Bs[128][40];

  const int tid  = threadIdx.x;
  const int lane = tid & 31;
  const int w    = tid >> 5;
  const int wm   = w & 1;          // 2 waves along M
  const int wn   = w >> 1;         // 4 waves along N
  const int bm   = blockIdx.y * 128;
  const int bn   = blockIdx.x * 128;

  const int ml = lane & 15;
  const int kb = (lane < 16) ? 0 : 8;        // A-frag K base (low VGPRs)
  const int kh = (lane >> 4) * 16;           // unused; B-frag also uses kb

  v8f acc[4][2];
#pragma unroll
  for (int i = 0; i < 4; ++i)
#pragma unroll
    for (int j = 0; j < 2; ++j) acc[i][j] = zero8();

  const int srow = tid >> 1;                 // staging row 0..127
  const int sseg = (tid & 1) * 16;           // staging col 0 / 16

  for (int k0 = 0; k0 < K; k0 += 32) {
    // ---- stage A tile (128 x 32) ----
    if (A_F32) {
      const float* Ag = (const float*)Ap + (size_t)(bm + srow) * K + k0 + sseg;
      if (k0 + 32 < K) __builtin_prefetch(Ag + 32, 0, 1);
      v4f f0 = *(const v4f*)(Ag + 0);
      v4f f1 = *(const v4f*)(Ag + 4);
      v4f f2 = *(const v4f*)(Ag + 8);
      v4f f3 = *(const v4f*)(Ag + 12);
      v8bf p0, p1;
#pragma unroll
      for (int i = 0; i < 4; ++i) {
        p0[i]     = (bf16_t)f0[i];
        p0[i + 4] = (bf16_t)f1[i];
        p1[i]     = (bf16_t)f2[i];
        p1[i + 4] = (bf16_t)f3[i];
      }
      *(v8bf*)&As[srow][sseg]     = p0;
      *(v8bf*)&As[srow][sseg + 8] = p1;
    } else {
      const bf16_t* Ag = (const bf16_t*)Ap + (size_t)(bm + srow) * K + k0 + sseg;
      if (k0 + 32 < K) __builtin_prefetch(Ag + 32, 0, 1);
      *(v8bf*)&As[srow][sseg]     = *(const v8bf*)Ag;
      *(v8bf*)&As[srow][sseg + 8] = *(const v8bf*)(Ag + 8);
    }
    // ---- stage B tile (W rows: 128 x 32, already bf16) ----
    {
      const bf16_t* Wg = W + (size_t)(bn + srow) * K + k0 + sseg;
      if (k0 + 32 < K) __builtin_prefetch(Wg + 32, 0, 1);
      *(v8bf*)&Bs[srow][sseg]     = *(const v8bf*)Wg;
      *(v8bf*)&Bs[srow][sseg + 8] = *(const v8bf*)(Wg + 8);
    }
    __syncthreads();

    // ---- fragments + WMMA ----
    v16bf a[4], b[2];
#pragma unroll
    for (int i = 0; i < 4; ++i) {
      const bf16_t* p = &As[wm * 64 + i * 16 + ml][0];
      a[i] = cat8(*(const v8bf*)(p + kb), *(const v8bf*)(p + kb + 16));
    }
#pragma unroll
    for (int j = 0; j < 2; ++j) {
      const bf16_t* p = &Bs[wn * 32 + j * 16 + ml][0];
      b[j] = cat8(*(const v8bf*)(p + kb), *(const v8bf*)(p + kb + 16));
    }
#pragma unroll
    for (int i = 0; i < 4; ++i)
#pragma unroll
      for (int j = 0; j < 2; ++j) acc[i][j] = wmma_bf16(a[i], b[j], acc[i][j]);

    __syncthreads();
  }

  // ---- epilogue: bias + scale + store ----
#pragma unroll
  for (int j = 0; j < 2; ++j) {
    const int gcol = bn + wn * 32 + j * 16 + ml;
    const float bv = bias[gcol];
#pragma unroll
    for (int i = 0; i < 4; ++i) {
      const int rowbase = bm + wm * 64 + i * 16 + ((lane >> 4) << 3);
      if (OUT_MODE == 2) {
        v8bf pk;
#pragma unroll
        for (int r = 0; r < 8; ++r)
          pk[r] = (bf16_t)((acc[i][j][r] + bv) * outScale);
        *(v8bf*)((bf16_t*)Outp + (size_t)gcol * M + rowbase) = pk;
      } else {
#pragma unroll
        for (int r = 0; r < 8; ++r) {
          const size_t gm  = (size_t)(rowbase + r);
          const float  val = (acc[i][j][r] + bv) * outScale;
          if (OUT_MODE == 1) ((float*)Outp)[gm * N + gcol]  = val;
          else               ((bf16_t*)Outp)[gm * N + gcol] = (bf16_t)val;
        }
      }
    }
  }
}

// ---------------------------------------------------------------------------
// Flash attention: grid (S/128, B*H), 256 threads = 8 waves.
// Double-buffered K/V tiles; wave 0 DMAs tile i+1 via the Tensor Data Mover
// (TENSOR_LOAD_TO_LDS + s_wait_tensorcnt) while all waves compute on tile i.
// Q pre-scaled by 0.125*log2(e); softmax in base-2; DPP row-max;
// row-sum via P @ ones WMMA.  V arrives pre-transposed column-major.
// ---------------------------------------------------------------------------
__global__ __launch_bounds__(256) void flash_attn_kernel(
    const bf16_t* __restrict__ Q, const bf16_t* __restrict__ Kp,
    const bf16_t* __restrict__ Vt, bf16_t* __restrict__ Oc) {
  __shared__ bf16_t Ks[2][32][72];   // 32 keys x hd=64 (+8 pad = 16B)
  __shared__ bf16_t Vts[2][64][40];  // hd x 32 keys (+8 pad = 16B)
  __shared__ bf16_t Ps[8][16][40];   // per-wave P tile 16x32 (+pad)

  const int tid  = threadIdx.x;
  const int lane = tid & 31;
  const int w    = tid >> 5;
  const int b    = blockIdx.y >> 4;
  const int h    = blockIdx.y & 15;
  const size_t rowbase = (size_t)b * SEQ + blockIdx.x * 128;
  const int    colbase = h * HDIM;

  const int ml = lane & 15;
  const int kb = (lane < 16) ? 0 : 8;
  const int kh = (lane >> 4) * 16;
  const int rofs = (lane >> 4) << 3;   // C-frag row offset for this half-wave

  // Q fragments (held in registers for the whole kernel)
  v16bf qa[2];
  {
    const bf16_t* qp = Q + (rowbase + w * 16 + ml) * MODEL_D + colbase;
#pragma unroll
    for (int s = 0; s < 2; ++s)
      qa[s] = cat8(*(const v8bf*)(qp + s * 32 + kb),
                   *(const v8bf*)(qp + s * 32 + kb + 16));
  }

  // all-ones B fragment for row-sum-by-WMMA
  v16bf ones;
#pragma unroll
  for (int i = 0; i < 16; ++i) ones[i] = (bf16_t)1.0f;

  float mi[8], li[8];
#pragma unroll
  for (int r = 0; r < 8; ++r) { mi[r] = -__builtin_inff(); li[r] = 0.0f; }
  v8f o[4];
#pragma unroll
  for (int j = 0; j < 4; ++j) o[j] = zero8();

  const int NIT = SEQ / 32;

#if USE_TDM
  // Prologue: DMA tile 0 into buffer 0 (wave 0 only; TDM ignores EXEC).
  if (w == 0) {
    const bf16_t* kg = Kp + ((size_t)b * SEQ + 0) * MODEL_D + colbase;
    tdm_load_2d((unsigned)(uintptr_t)&Ks[0][0][0], kg, HDIM, 32, MODEL_D,
                /*pad: 32dw*/ 4, /*4dw*/ 3);
    const bf16_t* vg = Vt + (size_t)colbase * MROWS + (size_t)b * SEQ;
    tdm_load_2d((unsigned)(uintptr_t)&Vts[0][0][0], vg, 32, HDIM, MROWS,
                /*pad: 16dw*/ 3, /*4dw*/ 3);
  }
#else
  const int krow = tid >> 3, kseg = (tid & 7) * 8;   // K staging coords
  const int vc = tid >> 2, vseg = (tid & 3) * 8;     // V^T staging coords
#endif

  for (int it = 0; it < NIT; ++it) {
    const int buf = it & 1;
    const int kb0 = it * 32;

#if USE_TDM
    if (w == 0) {
      if (it + 1 < NIT) {
        // DMA next tile into the other buffer (its readers finished at the
        // end-of-iteration barrier of it-1), then wait for the current tile:
        // TENSORcnt is per-wave and in-order, so <=2 means tile `it` landed.
        const int nb0 = kb0 + 32;
        const bf16_t* kg = Kp + ((size_t)b * SEQ + nb0) * MODEL_D + colbase;
        tdm_load_2d((unsigned)(uintptr_t)&Ks[buf ^ 1][0][0], kg, HDIM, 32,
                    MODEL_D, 4, 3);
        const bf16_t* vg =
            Vt + (size_t)colbase * MROWS + (size_t)b * SEQ + nb0;
        tdm_load_2d((unsigned)(uintptr_t)&Vts[buf ^ 1][0][0], vg, 32, HDIM,
                    MROWS, 3, 3);
        __builtin_amdgcn_s_wait_tensorcnt(2);
      } else {
        __builtin_amdgcn_s_wait_tensorcnt(0);
      }
    }
#else
    {
      const bf16_t* src =
          Kp + ((size_t)b * SEQ + kb0 + krow) * MODEL_D + colbase + kseg;
      *(v8bf*)&Ks[buf][krow][kseg] = *(const v8bf*)src;
    }
    {
      const bf16_t* src =
          Vt + (size_t)(colbase + vc) * MROWS + (size_t)b * SEQ + kb0 + vseg;
      *(v8bf*)&Vts[buf][vc][vseg] = *(const v8bf*)src;
    }
#endif
    __syncthreads();

    // ---- scores S = Q K^T (pre-scaled): two 16x16 C-frags ----
    v8f sc[2];
#pragma unroll
    for (int kt = 0; kt < 2; ++kt) {
      v8f c = zero8();
#pragma unroll
      for (int s = 0; s < 2; ++s) {
        const bf16_t* p = &Ks[buf][kt * 16 + ml][s * 32 + kb];
        v16bf bk = cat8(*(const v8bf*)p, *(const v8bf*)(p + 16));
        c = wmma_bf16(qa[s], bk, c);
      }
      sc[kt] = c;
    }

    // ---- online softmax: DPP row-max, exp2, P -> LDS (C layout) ----
    float alpha[8];
#pragma unroll
    for (int r = 0; r < 8; ++r) {
      const float s0 = sc[0][r];
      const float s1 = sc[1][r];
      const float rowmax = rowmax16(fmax_raw(s0, s1));
      const float mnew = fmax_raw(mi[r], rowmax);
      alpha[r] = __builtin_amdgcn_exp2f(mi[r] - mnew);
      mi[r] = mnew;
      const float p0 = __builtin_amdgcn_exp2f(s0 - mnew);
      const float p1 = __builtin_amdgcn_exp2f(s1 - mnew);
      const int prow = r + rofs;                 // C-layout row
      Ps[w][prow][ml]      = (bf16_t)p0;
      Ps[w][prow][ml + 16] = (bf16_t)p1;
    }
    __syncthreads();   // make P tile safely visible for A-layout reload

    // ---- reload P as A-fragment ----
    v16bf pa;
    {
      const bf16_t* p = &Ps[w][ml][0];
      pa = cat8(*(const v8bf*)(p + kb), *(const v8bf*)(p + kb + 16));
    }

    // ---- row sums in one WMMA: rs[r] = sum_k P[row, k] ----
    v8f rs = wmma_bf16(pa, ones, zero8());
#pragma unroll
    for (int r = 0; r < 8; ++r) {
      li[r] = li[r] * alpha[r] + rs[r];
#pragma unroll
      for (int j = 0; j < 4; ++j) o[j][r] = o[j][r] * alpha[r];
    }

    // ---- O += P V ----
#pragma unroll
    for (int j = 0; j < 4; ++j) {
      const bf16_t* p = &Vts[buf][j * 16 + ml][kb];
      v16bf vb = cat8(*(const v8bf*)p, *(const v8bf*)(p + 16));
      o[j] = wmma_bf16(pa, vb, o[j]);
    }
    __syncthreads();   // readers done before this buffer is overwritten
  }

  // ---- finalize: O /= l, write into concat layout [B*S, D] ----
#pragma unroll
  for (int r = 0; r < 8; ++r) {
    const float inv = __builtin_amdgcn_rcpf(li[r]);
    const size_t gm = rowbase + w * 16 + r + rofs;
#pragma unroll
    for (int j = 0; j < 4; ++j) {
      const int col = colbase + j * 16 + ml;
      Oc[gm * MODEL_D + col] = (bf16_t)(o[j][r] * inv);
    }
  }
}

// ---------------------------------------------------------------------------
// Host-side launcher
// ---------------------------------------------------------------------------
extern "C" void kernel_launch(void* const* d_in, const int* in_sizes, int n_in,
                              void* d_out, int out_size, void* d_ws,
                              size_t ws_size, hipStream_t stream) {
  (void)in_sizes; (void)n_in; (void)out_size; (void)ws_size;
  const float* q  = (const float*)d_in[0];
  const float* k  = (const float*)d_in[1];
  const float* v  = (const float*)d_in[2];
  const float* Wq = (const float*)d_in[3];
  const float* bq = (const float*)d_in[4];
  const float* Wo = (const float*)d_in[5];
  const float* bo = (const float*)d_in[6];
  float* out = (float*)d_out;

  const size_t actElems = (size_t)MROWS * MODEL_D;   // 8M bf16 elems each
  bf16_t* qp   = (bf16_t*)d_ws;
  bf16_t* kp   = qp + actElems;
  bf16_t* vt   = kp + actElems;        // column-major [MODEL_D, MROWS]
  bf16_t* conc = vt + actElems;
  bf16_t* WqB  = conc + actElems;
  bf16_t* WoB  = WqB + (size_t)MODEL_D * MODEL_D;
  // total workspace: 4*16MB + 2*2MB = 68MB

  const int wn4 = (MODEL_D * MODEL_D) / 4;
  cvt_f32_bf16_kernel<<<(wn4 + 255) / 256, 256, 0, stream>>>(Wq, WqB, wn4);
  cvt_f32_bf16_kernel<<<(wn4 + 255) / 256, 256, 0, stream>>>(Wo, WoB, wn4);

  dim3 gg(MODEL_D / 128, MROWS / 128);   // (8, 64)
  // q projection: fold 1/sqrt(hd)*log2(e) so softmax is pure base-2.
  gemm_wmma_kernel<true, 0><<<gg, 256, 0, stream>>>(
      q, WqB, bq, qp, MROWS, MODEL_D, MODEL_D, QSCALE);
  gemm_wmma_kernel<true, 0><<<gg, 256, 0, stream>>>(
      k, WqB, bq, kp, MROWS, MODEL_D, MODEL_D, 1.0f);
  // v projection: store column-major (transposed) for the flash PV stage.
  gemm_wmma_kernel<true, 2><<<gg, 256, 0, stream>>>(
      v, WqB, bq, vt, MROWS, MODEL_D, MODEL_D, 1.0f);

  flash_attn_kernel<<<dim3(SEQ / 128, BATCH * HEADS), 256, 0, stream>>>(
      qp, kp, vt, conc);

  gemm_wmma_kernel<false, 1><<<gg, 256, 0, stream>>>(
      conc, WoB, bo, out, MROWS, MODEL_D, MODEL_D, 1.0f);
}
